// MultiheadAttention_66881230733381
// MI455X (gfx1250) — matrix operand order repaired and checked by reading, not compiled, not run on hardware
//
#include <hip/hip_runtime.h>
#include <stdint.h>

// ---------------------------------------------------------------------------
// MI455X (gfx1250) multi-head attention: bf16 WMMA GEMMs + flash attention.
// B=4, S=T=2048, H=A=1024, NH=16, HD=64.  wave32, 16x16x32 bf16 WMMA.
// Attention computes S^T = K @ Q^T so the softmax reduction is in-lane and
// the softmax state is one scalar per lane; O^T = Vt @ P^T.
// Double-buffered 64-deep global->LDS async tiles hide load latency.
// ---------------------------------------------------------------------------

typedef __attribute__((ext_vector_type(16))) __bf16 v16bf;
typedef __attribute__((ext_vector_type(8)))  __bf16 v8bf;
typedef __attribute__((ext_vector_type(4)))  __bf16 v4bf;
typedef __attribute__((ext_vector_type(8)))  float  v8f;

#define DEV __device__ __forceinline__

DEV v16bf pack16(v8bf lo, v8bf hi) {
  v16bf r;
#pragma unroll
  for (int i = 0; i < 8; ++i) { r[i] = lo[i]; r[i + 8] = hi[i]; }
  return r;
}

// LDS byte offset of a __shared__ object: generic LDS addresses carry the
// wave-relative LDS offset in the low 32 bits (ISA 10.2 aperture rules).
DEV uint32_t lds_off(const void* p) { return (uint32_t)(uintptr_t)p; }

// Async global -> LDS, 16 bytes per lane (GVS mode: saddr64 + vaddr32 offset).
DEV void async_b128(uint32_t lds, uint32_t voff, uint64_t base) {
  asm volatile("global_load_async_to_lds_b128 %0, %1, %2"
               :: "v"(lds), "v"(voff), "s"(base) : "memory");
}
DEV void wait_async0() { asm volatile("s_wait_asynccnt 0" ::: "memory"); }

DEV v8f wmma_bf16(v16bf a, v16bf b, v8f c) {
  return __builtin_amdgcn_wmma_f32_16x16x32_bf16(false, a, false, b,
                                                 (short)0, c, false, false);
}

// ---------------------------------------------------------------------------
// fp32 -> bf16 elementwise (float4 / 4x bf16 per thread)
// ---------------------------------------------------------------------------
__global__ __launch_bounds__(256) void cvt_bf16(const float4* __restrict__ in,
                                                v4bf* __restrict__ out) {
  int i = blockIdx.x * 256 + threadIdx.x;
  float4 v = in[i];
  v4bf o;
  o[0] = (__bf16)v.x; o[1] = (__bf16)v.y; o[2] = (__bf16)v.z; o[3] = (__bf16)v.w;
  out[i] = o;
}

// 1024x1024 transpose + convert: out[n*1024+k] = (bf16) in[k*1024+n]
__global__ __launch_bounds__(256) void transpose_cvt(const float* __restrict__ in,
                                                     __bf16* __restrict__ out) {
  int idx = blockIdx.x * 256 + threadIdx.x;
  int n = idx >> 10, k = idx & 1023;
  out[idx] = (__bf16)in[k * 1024 + n];
}

// ---------------------------------------------------------------------------
// GEMM: C[M=8192][N=1024] = (A[M][K=1024](bf16) @ Bt^T + bias) * scale
//   Bt is N x K bf16 (pre-transposed weights).
//   MODE 0: write bf16, Q/K head layout  [b][h][pos][hd]
//   MODE 1: write bf16, V transposed     [b][h][hd][pos]  (packed stores)
//   MODE 2: write fp32 row-major (final output)
// Block: 256 thr (8 waves, 2x4), tile 64(m) x 128(n), wave tile 32x32,
// K-step 64 (8 WMMA / wave / iter), double-buffered async global->LDS.
// ---------------------------------------------------------------------------
template <int MODE>
__global__ __launch_bounds__(256)
void gemm_wmma(const __bf16* __restrict__ Aa, const __bf16* __restrict__ Bt,
               const float* __restrict__ bias, float scale,
               void* __restrict__ outp) {
  __shared__ __align__(16) __bf16 As[2][64 * 72];    // 64 rows, stride 72
  __shared__ __align__(16) __bf16 Bs[2][128 * 72];   // 128 rows, stride 72

  const int tid = threadIdx.x;
  const int lane = tid & 31, wave = tid >> 5;
  const int wm = wave >> 2, wn = wave & 3;
  const int m_base = blockIdx.y * 64;
  const int n_base = blockIdx.x * 128;
  const int lm = lane & 15, lh = lane >> 4;

  v8f acc[2][2];
#pragma unroll
  for (int i = 0; i < 2; ++i)
#pragma unroll
    for (int j = 0; j < 2; ++j)
#pragma unroll
      for (int e = 0; e < 8; ++e) acc[i][j][e] = 0.0f;

  // async-copy geometry: A tile 64x64 bf16 = 512 x 16B chunks (2/thread),
  //                      B tile 128x64 bf16 = 1024 chunks (4/thread)
  uint32_t ldsA[2][2], gAo[2], ldsB[2][4], gBo[4];
#pragma unroll
  for (int c = 0; c < 2; ++c) {
    int cc = tid + c * 256, r = cc >> 3, ch = cc & 7;
    ldsA[0][c] = lds_off(As[0]) + (uint32_t)(r * 144 + ch * 16);
    ldsA[1][c] = lds_off(As[1]) + (uint32_t)(r * 144 + ch * 16);
    gAo[c] = ((uint32_t)(m_base + r) * 1024u + (uint32_t)ch * 8u) * 2u;
  }
#pragma unroll
  for (int c = 0; c < 4; ++c) {
    int cc = tid + c * 256, r = cc >> 3, ch = cc & 7;
    ldsB[0][c] = lds_off(Bs[0]) + (uint32_t)(r * 144 + ch * 16);
    ldsB[1][c] = lds_off(Bs[1]) + (uint32_t)(r * 144 + ch * 16);
    gBo[c] = ((uint32_t)(n_base + r) * 1024u + (uint32_t)ch * 8u) * 2u;
  }
  const uint64_t baseA = (uint64_t)(uintptr_t)Aa;
  const uint64_t baseB = (uint64_t)(uintptr_t)Bt;

  auto issue = [&](int buf, int k0) {
    const uint32_t kb = (uint32_t)k0 * 2u;
#pragma unroll
    for (int c = 0; c < 2; ++c) async_b128(ldsA[buf][c], gAo[c] + kb, baseA);
#pragma unroll
    for (int c = 0; c < 4; ++c) async_b128(ldsB[buf][c], gBo[c] + kb, baseB);
  };

  issue(0, 0);
  for (int it = 0; it < 16; ++it) {
    wait_async0();          // this buffer's tiles landed (issued last iter)
    __syncthreads();        // everyone's data in; previous compute finished
    if (it + 1 < 16) issue((it + 1) & 1, (it + 1) * 64);

    const __bf16* as_ = As[it & 1];
    const __bf16* bs_ = Bs[it & 1];
    v16bf af[2][2], bf2[2][2];
#pragma unroll
    for (int ks = 0; ks < 2; ++ks) {
#pragma unroll
      for (int i2 = 0; i2 < 2; ++i2) {
        int row = wm * 32 + i2 * 16 + lm;
        v8bf lo = *(const v8bf*)&as_[row * 72 + ks * 32 + lh * 8];
        v8bf hi = *(const v8bf*)&as_[row * 72 + ks * 32 + 16 + lh * 8];
        af[ks][i2] = pack16(lo, hi);
      }
#pragma unroll
      for (int j2 = 0; j2 < 2; ++j2) {
        int row = wn * 32 + j2 * 16 + lm;
        v8bf lo = *(const v8bf*)&bs_[row * 72 + ks * 32 + lh * 8];
        v8bf hi = *(const v8bf*)&bs_[row * 72 + ks * 32 + 16 + lh * 8];
        bf2[ks][j2] = pack16(lo, hi);
      }
    }
#pragma unroll
    for (int i2 = 0; i2 < 2; ++i2)
#pragma unroll
      for (int j2 = 0; j2 < 2; ++j2) {
        acc[i2][j2] = wmma_bf16(af[0][i2], bf2[0][j2], acc[i2][j2]);
        acc[i2][j2] = wmma_bf16(af[1][i2], bf2[1][j2], acc[i2][j2]);
      }
  }

  // epilogue
#pragma unroll
  for (int i2 = 0; i2 < 2; ++i2) {
#pragma unroll
    for (int j2 = 0; j2 < 2; ++j2) {
      int c = n_base + wn * 32 + j2 * 16 + lm;
      float bv = bias[c];
      int r0 = m_base + wm * 32 + i2 * 16 + 8 * lh;   // 8 consecutive rows
      if (MODE == 1) {
        int bidx = r0 >> 11, pos0 = r0 & 2047;
        int hh = c >> 6, dd = c & 63;
        size_t idx = (((size_t)bidx * 16 + hh) * 64 + dd) * 2048 + pos0;
        v8bf ov;
#pragma unroll
        for (int e = 0; e < 8; ++e)
          ov[e] = (__bf16)((acc[i2][j2][e] + bv) * scale);
        *(v8bf*)&((__bf16*)outp)[idx] = ov;
      } else {
#pragma unroll
        for (int e = 0; e < 8; ++e) {
          int r = r0 + e;
          float val = (acc[i2][j2][e] + bv) * scale;
          if (MODE == 2) {
            ((float*)outp)[(size_t)r * 1024 + c] = val;
          } else {
            int bidx = r >> 11, pos = r & 2047;
            int hh = c >> 6, dd = c & 63;
            size_t idx = (((size_t)bidx * 16 + hh) * 2048 + pos) * 64 + dd;
            ((__bf16*)outp)[idx] = (__bf16)val;
          }
        }
      }
    }
  }
}

// ---------------------------------------------------------------------------
// Flash attention (transposed-score form).  Grid: (S/128, B*NH).
// Block = 8 waves, wave = 16 queries (one lane column per query).
// Q: [b,h,s,hd] bf16 (pre-scaled by 1/8);  K: [b,h,t,hd];  Vt: [b,h,hd,t];
// O: [b*S][A] bf16 row-major.   Key-tile = 64 (16 WMMA / wave / iter).
//   S^T tile: rows = 64 keys (A = K tile), cols = 16 queries (B = Q^T).
//   Softmax over keys = in-lane reduce (32 vals) + one xor-16 shuffle.
//   O^T = Vt_tile (A) @ P^T (B); P^T staged as packed LDS rows per query.
// ---------------------------------------------------------------------------
__global__ __launch_bounds__(256)
void flash_attn(const __bf16* __restrict__ Q, const __bf16* __restrict__ K,
                const __bf16* __restrict__ Vt,
                const unsigned char* __restrict__ mask,
                __bf16* __restrict__ O) {
  __shared__ __align__(16) __bf16 Ks[2][64 * 72];    // 64 keys x 64 hd (pad 72)
  __shared__ __align__(16) __bf16 Vs[2][64 * 72];    // 64 hd x 64 keys (pad 72)
  __shared__ __align__(16) __bf16 Ps[8][16 * 72];    // per-wave P^T: 16 q x 64 t

  const int tid = threadIdx.x, lane = tid & 31, wave = tid >> 5;
  const int lm = lane & 15, lh = lane >> 4;
  const int bh = blockIdx.y;               // b*16 + h
  const int b = bh >> 4, h = bh & 15;
  const int s0 = blockIdx.x * 128 + wave * 16;

  // Q as B-fragments: lane column = query s0+lm, k-steps {hd 0..31},{32..63}
  v16bf bq[2];
  {
    const size_t qrow = ((size_t)bh * 2048 + (size_t)(s0 + lm)) * 64;
#pragma unroll
    for (int ks = 0; ks < 2; ++ks) {
      v8bf lo = *(const v8bf*)&Q[qrow + ks * 32 + lh * 8];
      v8bf hi = *(const v8bf*)&Q[qrow + ks * 32 + 16 + lh * 8];
      bq[ks] = pack16(lo, hi);
    }
  }

  float mrun = -3.0e38f, lrun = 0.0f;     // per-lane (per-query) softmax state
  v8f o[4];
#pragma unroll
  for (int f = 0; f < 4; ++f)
#pragma unroll
    for (int e = 0; e < 8; ++e) o[f][e] = 0.0f;

  // async-copy geometry: K tile 64x64 = 512 x 16B chunks (2/thread), V same
  uint32_t ldsK[2][2], gKo[2], ldsV[2][2], gVo[2];
#pragma unroll
  for (int c = 0; c < 2; ++c) {
    int cc = tid + c * 256, r = cc >> 3, ch = cc & 7;
    ldsK[0][c] = lds_off(Ks[0]) + (uint32_t)(r * 144 + ch * 16);
    ldsK[1][c] = lds_off(Ks[1]) + (uint32_t)(r * 144 + ch * 16);
    gKo[c] = (((uint32_t)bh * 2048u + (uint32_t)r) * 64u + (uint32_t)ch * 8u) * 2u;
    ldsV[0][c] = lds_off(Vs[0]) + (uint32_t)(r * 144 + ch * 16);
    ldsV[1][c] = lds_off(Vs[1]) + (uint32_t)(r * 144 + ch * 16);
    gVo[c] = (((uint32_t)bh * 64u + (uint32_t)r) * 2048u + (uint32_t)ch * 8u) * 2u;
  }
  const uint64_t baseK = (uint64_t)(uintptr_t)K;
  const uint64_t baseV = (uint64_t)(uintptr_t)Vt;
  const unsigned char* mrow = mask + (size_t)b * 2048;
  __bf16* pw = Ps[wave];

  auto issue = [&](int buf, int t) {
#pragma unroll
    for (int c = 0; c < 2; ++c) {
      async_b128(ldsK[buf][c], gKo[c] + (uint32_t)t * 128u, baseK);
      async_b128(ldsV[buf][c], gVo[c] + (uint32_t)t * 2u, baseV);
    }
  };

  issue(0, 0);
  for (int it = 0; it < 32; ++it) {
    const int t = it * 64;
    wait_async0();
    __syncthreads();
    if (it + 1 < 32) issue((it + 1) & 1, t + 64);
    const __bf16* ks_ = Ks[it & 1];
    const __bf16* vs_ = Vs[it & 1];

    // --- S^T: four 16(key) x 16(query) tiles, contraction over hd=64 ---
    float p[4][8];
#pragma unroll
    for (int tt = 0; tt < 4; ++tt) {
      v8f s;
#pragma unroll
      for (int e = 0; e < 8; ++e) s[e] = 0.0f;
#pragma unroll
      for (int ks2 = 0; ks2 < 2; ++ks2) {
        int row = tt * 16 + lm;           // key row of A fragment
        v8bf lo = *(const v8bf*)&ks_[row * 72 + ks2 * 32 + lh * 8];
        v8bf hi = *(const v8bf*)&ks_[row * 72 + ks2 * 32 + 16 + lh * 8];
        s = wmma_bf16(pack16(lo, hi), bq[ks2], s);
      }
      // element e -> key t + tt*16 + 8*lh + e ; exact reference masking
      const uint32_t* mp = (const uint32_t*)(mrow + t + tt * 16 + lh * 8);
      uint32_t m0 = mp[0], m1 = mp[1];
#pragma unroll
      for (int e = 0; e < 8; ++e) {
        uint32_t byte = (e < 4) ? ((m0 >> (8 * e)) & 0xffu)
                                : ((m1 >> (8 * (e - 4))) & 0xffu);
        p[tt][e] = byte ? s[e] : -100000.0f;
      }
    }

    // --- online softmax over 64 keys: in-lane reduce + xor-16 shuffle ---
    float tmax = p[0][0];
#pragma unroll
    for (int tt = 0; tt < 4; ++tt)
#pragma unroll
      for (int e = 0; e < 8; ++e) tmax = fmaxf(tmax, p[tt][e]);
    tmax = fmaxf(tmax, __shfl_xor(tmax, 16, 32));
    float newm = fmaxf(mrun, tmax);
    float sc = __expf(mrun - newm);
    mrun = newm;
    float rs = 0.0f;
#pragma unroll
    for (int tt = 0; tt < 4; ++tt)
#pragma unroll
      for (int e = 0; e < 8; ++e) {
        float ev = __expf(p[tt][e] - newm);
        p[tt][e] = ev;
        rs += ev;
      }
    rs += __shfl_xor(rs, 16, 32);
    lrun = lrun * sc + rs;
#pragma unroll
    for (int f = 0; f < 4; ++f)
#pragma unroll
      for (int e = 0; e < 8; ++e) o[f][e] *= sc;

    // --- stage P^T (row per query, packed 16B stores) ---
#pragma unroll
    for (int tt = 0; tt < 4; ++tt) {
      v8bf pv;
#pragma unroll
      for (int e = 0; e < 8; ++e) pv[e] = (__bf16)p[tt][e];
      *(v8bf*)&pw[lm * 72 + tt * 16 + lh * 8] = pv;
    }
    // B fragments of P^T (same-wave DS ops are in-order)
    v16bf bp[2];
#pragma unroll
    for (int ks2 = 0; ks2 < 2; ++ks2) {
      v8bf lo = *(const v8bf*)&pw[lm * 72 + ks2 * 32 + lh * 8];
      v8bf hi = *(const v8bf*)&pw[lm * 72 + ks2 * 32 + 16 + lh * 8];
      bp[ks2] = pack16(lo, hi);
    }
    // --- O^T += Vt_tile @ P^T : 4 hd tiles x 2 k-steps ---
#pragma unroll
    for (int f = 0; f < 4; ++f) {
      int row = f * 16 + lm;              // hd row of A fragment
#pragma unroll
      for (int ks2 = 0; ks2 < 2; ++ks2) {
        v8bf lo = *(const v8bf*)&vs_[row * 72 + ks2 * 32 + lh * 8];
        v8bf hi = *(const v8bf*)&vs_[row * 72 + ks2 * 32 + 16 + lh * 8];
        o[f] = wmma_bf16(pack16(lo, hi), bp[ks2], o[f]);
      }
    }
  }

  // --- normalize + packed store: lane = query row, elements = contiguous hd ---
  const float inv = 1.0f / lrun;
  const size_t r = (size_t)b * 2048 + (size_t)blockIdx.x * 128 + wave * 16 + lm;
#pragma unroll
  for (int f = 0; f < 4; ++f) {
    v8bf ov;
#pragma unroll
    for (int e = 0; e < 8; ++e) ov[e] = (__bf16)(o[f][e] * inv);
    *(v8bf*)&O[r * 1024 + (size_t)h * 64 + f * 16 + lh * 8] = ov;
  }
}

// ---------------------------------------------------------------------------
// host launcher
// ---------------------------------------------------------------------------
extern "C" void kernel_launch(void* const* d_in, const int* in_sizes, int n_in,
                              void* d_out, int out_size, void* d_ws,
                              size_t ws_size, hipStream_t stream) {
  (void)in_sizes; (void)n_in; (void)out_size; (void)ws_size;
  const float* x  = (const float*)d_in[0];
  const float* y  = (const float*)d_in[1];
  const unsigned char* mask = (const unsigned char*)d_in[2];  // bool -> 1 byte
  const float* wq = (const float*)d_in[3];
  const float* bq = (const float*)d_in[4];
  const float* wk = (const float*)d_in[5];
  const float* bk = (const float*)d_in[6];
  const float* wv = (const float*)d_in[7];
  const float* bv = (const float*)d_in[8];
  const float* wo = (const float*)d_in[9];
  const float* bo = (const float*)d_in[10];

  char* w = (char*)d_ws;
  auto alloc = [&](size_t bytes) {
    void* p = (void*)w;
    w += (bytes + 255) & ~(size_t)255;
    return p;
  };
  const size_t MB16 = 8192ull * 1024 * 2;   // 16 MiB (8192x1024 bf16)
  const size_t MB2  = 1024ull * 1024 * 2;   // 2 MiB  (1024x1024 bf16)
  __bf16* xb  = (__bf16*)alloc(MB16);
  __bf16* yb  = (__bf16*)alloc(MB16);
  __bf16* wqT = (__bf16*)alloc(MB2);
  __bf16* wkT = (__bf16*)alloc(MB2);
  __bf16* wvT = (__bf16*)alloc(MB2);
  __bf16* woT = (__bf16*)alloc(MB2);
  __bf16* Qb  = (__bf16*)alloc(MB16);       // [b,h,s,hd], pre-scaled by 1/8
  __bf16* Kb  = (__bf16*)alloc(MB16);       // [b,h,t,hd]
  __bf16* Vtb = (__bf16*)alloc(MB16);       // [b,h,hd,t]
  __bf16* Ob  = (__bf16*)alloc(MB16);       // [b*S][A]

  cvt_bf16<<<8192, 256, 0, stream>>>((const float4*)x, (v4bf*)xb);
  cvt_bf16<<<8192, 256, 0, stream>>>((const float4*)y, (v4bf*)yb);
  transpose_cvt<<<4096, 256, 0, stream>>>(wq, wqT);
  transpose_cvt<<<4096, 256, 0, stream>>>(wk, wkT);
  transpose_cvt<<<4096, 256, 0, stream>>>(wv, wvT);
  transpose_cvt<<<4096, 256, 0, stream>>>(wo, woT);

  dim3 gg(8, 128);    // N/128, M/64
  gemm_wmma<0><<<gg, 256, 0, stream>>>(xb, wqT, bq, 0.125f, Qb);  // Q, pre-scaled
  gemm_wmma<0><<<gg, 256, 0, stream>>>(yb, wkT, bk, 1.0f, Kb);
  gemm_wmma<1><<<gg, 256, 0, stream>>>(yb, wvT, bv, 1.0f, Vtb);

  flash_attn<<<dim3(16, 64), 256, 0, stream>>>(Qb, Kb, Vtb, mask, Ob);

  gemm_wmma<2><<<gg, 256, 0, stream>>>(Ob, woT, bo, 1.0f, d_out);
}
